// Seq2Seq_35124242547152
// MI455X (gfx1250) — matrix operand order repaired and checked by reading, not compiled, not run on hardware
//
#include <hip/hip_runtime.h>
#include <limits.h>
#include <math.h>

// ---------------------------------------------------------------------------
// Seq2Seq attention-LSTM decoder for MI455X (gfx1250, wave32, WMMA).
// B=32, T_ENC=512, ENC_H=E=H=512, V=32000, STEPS=64.
// bf16 weights cached in L2 (40MB < 192MB); all GEMMs via
// v_wmma_f32_16x16x32_bf16, one wave per N-tile computing BOTH 16-row
// M-tiles so each weight fragment load feeds two WMMAs.
// ---------------------------------------------------------------------------

typedef __bf16 bf16;
typedef __attribute__((ext_vector_type(16))) __bf16 v16bf;
typedef __attribute__((ext_vector_type(8)))  float  v8f;

static __device__ __forceinline__ v16bf make_v16(uint4 lo, uint4 hi) {
  union { uint4 u[2]; v16bf v; } x;
  x.u[0] = lo; x.u[1] = hi;
  return x.v;
}

// ------------------------- fp32 -> bf16 convert ----------------------------
__global__ void f2bf_kernel(const float* __restrict__ src, bf16* __restrict__ dst, int n) {
  int i = blockIdx.x * 256 + threadIdx.x;
  if (i < n) dst[i] = (bf16)src[i];
}

__global__ void init_mask_kernel(int* mask) {
  if (threadIdx.x < 32) mask[threadIdx.x] = 1;
}

// ------------------------- generic 32xN bf16 WMMA GEMM ---------------------
// out[32,N] (+ optional bf16 copy) = A[32,K](stride astride) @ W[N,K]^T + bias
// One wave per 16-col N-tile; computes both M-tiles (rows 0..15 and 16..31)
// so each weight fragment is loaded once and used by two WMMAs.
__global__ void gemm_bf16_32(const bf16* __restrict__ A, int astride,
                             const bf16* __restrict__ W,
                             const float* __restrict__ bias,
                             float* __restrict__ out,
                             bf16* __restrict__ out_bf, int obstride,
                             int K, int N, int do_relu, int accumulate) {
  int lane = threadIdx.x & 31;
  int wid  = threadIdx.x >> 5;
  int nt   = blockIdx.x * 4 + wid;
  int ntiles = N >> 4;
  if (nt >= ntiles) return;
  int n0 = nt * 16;
  int g    = lane >> 4;                 // lane half selects K-chunk per ISA layout
  int ncol = n0 + (lane & 15);

  const bf16* ap0 = A + (size_t)(lane & 15) * astride + g * 8;  // rows 0..15
  const bf16* ap1 = ap0 + (size_t)16 * astride;                 // rows 16..31
  const bf16* wp0 = W + (size_t)ncol * K + g * 8;               // B: same k chunks as A

  v8f acc0 = {0.f, 0.f, 0.f, 0.f, 0.f, 0.f, 0.f, 0.f};
  v8f acc1 = {0.f, 0.f, 0.f, 0.f, 0.f, 0.f, 0.f, 0.f};
  for (int k0 = 0; k0 < K; k0 += 32) {
    uint4 bl = *(const uint4*)(wp0 + k0);
    uint4 bh = *(const uint4*)(wp0 + k0 + 16);
    v16bf bv = make_v16(bl, bh);
    uint4 a0l = *(const uint4*)(ap0 + k0);
    uint4 a0h = *(const uint4*)(ap0 + k0 + 16);
    uint4 a1l = *(const uint4*)(ap1 + k0);
    uint4 a1h = *(const uint4*)(ap1 + k0 + 16);
    acc0 = __builtin_amdgcn_wmma_f32_16x16x32_bf16(
        false, make_v16(a0l, a0h), false, bv, (short)0, acc0, false, false);
    acc1 = __builtin_amdgcn_wmma_f32_16x16x32_bf16(
        false, make_v16(a1l, a1h), false, bv, (short)0, acc1, false, false);
  }

  float bval = bias ? bias[ncol] : 0.0f;
#pragma unroll
  for (int r = 0; r < 8; ++r) {
    int m = r + 8 * g;                  // C/D layout: vgpr r -> row r+8*(lane/16)
    float v0 = acc0[r] + bval;
    float v1 = acc1[r] + bval;
    if (accumulate) {
      v0 += out[(size_t)m * N + ncol];
      v1 += out[(size_t)(m + 16) * N + ncol];
    }
    if (do_relu) { v0 = fmaxf(v0, 0.0f); v1 = fmaxf(v1, 0.0f); }
    if (out) {
      out[(size_t)m * N + ncol]        = v0;
      out[(size_t)(m + 16) * N + ncol] = v1;
    }
    if (out_bf) {
      out_bf[(size_t)m * obstride + ncol]        = (bf16)v0;
      out_bf[(size_t)(m + 16) * obstride + ncol] = (bf16)v1;
    }
  }
}

// ------------------------- vocab-projection WMMA GEMM ----------------------
// logits[32,32000] = pre_bf[32,512] @ Wo_bf[32000,512]^T + bo
// A staged in LDS (padded stride, conflict-free ds_load_b128); one wave per
// N-tile computing both M-tiles; prefetch on the Wo stream.
#define SROW 520
__global__ void logits_gemm_kernel(const bf16* __restrict__ A,
                                   const bf16* __restrict__ W,
                                   const float* __restrict__ bias,
                                   float* __restrict__ out) {
  __shared__ bf16 sA[32 * SROW];
  int tid = threadIdx.x;
  for (int i = tid; i < 2048; i += 128) {        // 32 rows x 64 uint4 = 32KB
    int row = i >> 6, col = i & 63;
    ((uint4*)(sA + row * SROW))[col] = ((const uint4*)A)[i];
  }
  __syncthreads();

  int lane = tid & 31, wid = tid >> 5;
  int nt = blockIdx.x * 4 + wid;                 // 2000 N-tiles / 500 blocks
  int n0 = nt * 16;
  int g    = lane >> 4;
  int ncol = n0 + (lane & 15);

  const bf16* ap0 = sA + (lane & 15) * SROW + g * 8;
  const bf16* ap1 = ap0 + 16 * SROW;
  const bf16* wp0 = W + (size_t)ncol * 512 + g * 8;

  v8f acc0 = {0.f, 0.f, 0.f, 0.f, 0.f, 0.f, 0.f, 0.f};
  v8f acc1 = {0.f, 0.f, 0.f, 0.f, 0.f, 0.f, 0.f, 0.f};
#pragma unroll 4
  for (int k0 = 0; k0 < 512; k0 += 32) {
    __builtin_prefetch((const void*)(wp0 + k0 + 128), 0, 0); // global_prefetch_b8
    uint4 bl = *(const uint4*)(wp0 + k0);
    uint4 bh = *(const uint4*)(wp0 + k0 + 16);
    v16bf bv = make_v16(bl, bh);
    uint4 a0l = *(const uint4*)(ap0 + k0);
    uint4 a0h = *(const uint4*)(ap0 + k0 + 16);
    uint4 a1l = *(const uint4*)(ap1 + k0);
    uint4 a1h = *(const uint4*)(ap1 + k0 + 16);
    acc0 = __builtin_amdgcn_wmma_f32_16x16x32_bf16(
        false, make_v16(a0l, a0h), false, bv, (short)0, acc0, false, false);
    acc1 = __builtin_amdgcn_wmma_f32_16x16x32_bf16(
        false, make_v16(a1l, a1h), false, bv, (short)0, acc1, false, false);
  }

  float bval = bias[ncol];
#pragma unroll
  for (int r = 0; r < 8; ++r) {
    int m = r + 8 * g;
    out[(size_t)m * 32000 + ncol]        = acc0[r] + bval;
    out[(size_t)(m + 16) * 32000 + ncol] = acc1[r] + bval;
  }
}

// ------------------------- embedding gather --------------------------------
__global__ void emb_kernel(const float* __restrict__ embedding,
                           const int* __restrict__ trg,
                           bf16* __restrict__ cat1, bf16* __restrict__ cat2, int t) {
  int idx = blockIdx.x * 256 + threadIdx.x;      // 32*512 threads
  int b = idx >> 9, e = idx & 511;
  float val = 0.0f;
  if (t > 0) {
    int tok = trg[b * 64 + t - 1];
    val = embedding[(size_t)tok * 512 + e];
  }
  bf16 bv = (bf16)val;
  cat1[b * 1024 + e] = bv;
  cat2[b * 1024 + e] = bv;
}

// ------------------------- attention softmax (N=512, in place) -------------
__global__ void attn_softmax_kernel(float* __restrict__ s) {
  int b = blockIdx.x, tid = threadIdx.x;         // 256 threads, 2 elems each
  __shared__ float red[256];
  float x0 = s[b * 512 + tid], x1 = s[b * 512 + 256 + tid];
  red[tid] = fmaxf(x0, x1);
  __syncthreads();
  for (int off = 128; off > 0; off >>= 1) {
    if (tid < off) red[tid] = fmaxf(red[tid], red[tid + off]);
    __syncthreads();
  }
  float rm = red[0];
  __syncthreads();
  float e0 = __expf(x0 - rm), e1 = __expf(x1 - rm);
  red[tid] = e0 + e1;
  __syncthreads();
  for (int off = 128; off > 0; off >>= 1) {
    if (tid < off) red[tid] += red[tid + off];
    __syncthreads();
  }
  float inv = 1.0f / red[0];
  s[b * 512 + tid] = e0 * inv;
  s[b * 512 + 256 + tid] = e1 * inv;
}

// ------------------------- context: ctx[b,e] = sum_t aw[b,t]*enc[b,t,e] ----
__global__ void ctx_kernel(const float* __restrict__ aw,
                           const float* __restrict__ enc,
                           bf16* __restrict__ cat2) {
  int idx = blockIdx.x * 256 + threadIdx.x;      // 32*512 threads
  int b = idx >> 9, e = idx & 511;
  const float* a  = aw + b * 512;
  const float* ep = enc + (size_t)b * 512 * 512 + e;
  float acc = 0.0f;
  for (int t = 0; t < 512; ++t) acc += a[t] * ep[(size_t)t * 512];
  cat2[b * 1024 + 512 + e] = (bf16)acc;
}

// ------------------------- LSTM pointwise ----------------------------------
__global__ void lstm_kernel(const float* __restrict__ gates,
                            float* __restrict__ c,
                            bf16* __restrict__ cat1,
                            float* __restrict__ states, int t) {
  int idx = blockIdx.x * 256 + threadIdx.x;      // 32*512 threads
  int b = idx >> 9, j = idx & 511;
  const float* gr = gates + b * 2048;
  float i_ = gr[j], f_ = gr[512 + j], g_ = gr[1024 + j], o_ = gr[1536 + j];
  float si = 1.0f / (1.0f + __expf(-i_));
  float sf = 1.0f / (1.0f + __expf(-f_));
  float so = 1.0f / (1.0f + __expf(-o_));
  float cn = sf * c[b * 512 + j] + si * tanhf(g_);
  float hn = so * tanhf(cn);
  c[b * 512 + j] = cn;
  cat1[b * 1024 + 512 + j] = (bf16)hn;           // h_bf for next step / gates
  states[(size_t)b * 32768 + (size_t)t * 512 + j] = hn;
}

// ------------------------- log_softmax + argmax + mask ---------------------
__global__ void finalize_kernel(const float* __restrict__ logits,
                                int* __restrict__ mask,
                                float* __restrict__ lps,
                                float* __restrict__ preds,
                                float* __restrict__ masks, int t) {
  int b = blockIdx.x, tid = threadIdx.x;         // one block per batch row
  const float* row = logits + (size_t)b * 32000;
  __shared__ float red[256];
  __shared__ int sidx;
  float m = -INFINITY;
  for (int v = tid; v < 32000; v += 256) m = fmaxf(m, row[v]);
  red[tid] = m;
  __syncthreads();
  for (int off = 128; off > 0; off >>= 1) {
    if (tid < off) red[tid] = fmaxf(red[tid], red[tid + off]);
    __syncthreads();
  }
  float rm = red[0];
  __syncthreads();
  float s = 0.0f;
  int li = INT_MAX;
  for (int v = tid; v < 32000; v += 256) {
    float x = row[v];
    s += __expf(x - rm);
    if (x == rm && v < li) li = v;               // first occurrence of max
  }
  red[tid] = s;
  if (tid == 0) sidx = INT_MAX;
  __syncthreads();
  atomicMin(&sidx, li);
  for (int off = 128; off > 0; off >>= 1) {
    if (tid < off) red[tid] += red[tid + off];
    __syncthreads();
  }
  float lse = rm + logf(red[0]);
  __syncthreads();
  for (int v = tid; v < 32000; v += 256)
    lps[(size_t)b * 2048000 + (size_t)t * 32000 + v] = row[v] - lse;
  if (tid == 0) {
    int pred = sidx;
    preds[b * 64 + t] = (float)pred;
    int mo = mask[b];
    masks[b * 64 + t] = (float)mo;               // mask_out = mask before update
    mask[b] = (pred != 2 && mo) ? 1 : 0;         // EOS_IDX = 2
  }
}

// ---------------------------------------------------------------------------
extern "C" void kernel_launch(void* const* d_in, const int* in_sizes, int n_in,
                              void* d_out, int out_size, void* d_ws, size_t ws_size,
                              hipStream_t stream) {
  (void)in_sizes; (void)n_in; (void)out_size; (void)ws_size;

  const float* enc       = (const float*)d_in[0];   // [32,512,512]
  const float* enc_h     = (const float*)d_in[1];   // [1,32,512]
  const float* enc_c     = (const float*)d_in[2];   // [1,32,512]
  const int*   trg       = (const int*)d_in[3];     // [32,64]
  const float* embedding = (const float*)d_in[5];   // [32000,512]
  const float* Wa  = (const float*)d_in[6];
  const float* ba  = (const float*)d_in[7];
  const float* Wc  = (const float*)d_in[8];
  const float* bc  = (const float*)d_in[9];
  const float* Wh0 = (const float*)d_in[10];
  const float* Wc0 = (const float*)d_in[11];
  const float* Wih = (const float*)d_in[12];
  const float* Whh = (const float*)d_in[13];
  const float* bih = (const float*)d_in[14];
  const float* bhh = (const float*)d_in[15];
  const float* Wp  = (const float*)d_in[16];
  const float* bp  = (const float*)d_in[17];
  const float* Wo  = (const float*)d_in[18];
  const float* bo  = (const float*)d_in[19];

  // ---- workspace carve-up (256B aligned slots) ----
  char* p = (char*)d_ws;
  auto alloc = [&](size_t bytes) -> void* {
    void* r = (void*)p;
    p += (bytes + 255) & ~(size_t)255;
    return r;
  };
  bf16* wWa  = (bf16*)alloc(512 * 1024 * 2);
  bf16* wWc  = (bf16*)alloc(512 * 1024 * 2);
  bf16* wWih = (bf16*)alloc(2048 * 512 * 2);
  bf16* wWhh = (bf16*)alloc(2048 * 512 * 2);
  bf16* wWp  = (bf16*)alloc(512 * 512 * 2);
  bf16* wWh0 = (bf16*)alloc(512 * 512 * 2);
  bf16* wWc0 = (bf16*)alloc(512 * 512 * 2);
  bf16* wWo  = (bf16*)alloc((size_t)32000 * 512 * 2);
  bf16* ehbf = (bf16*)alloc(32 * 512 * 2);
  bf16* ecbf = (bf16*)alloc(32 * 512 * 2);
  bf16* cat1 = (bf16*)alloc(32 * 1024 * 2);       // [emb | h] bf16
  bf16* cat2 = (bf16*)alloc(32 * 1024 * 2);       // [emb | ctx] bf16
  bf16* combb= (bf16*)alloc(32 * 512 * 2);
  bf16* prebf= (bf16*)alloc(32 * 512 * 2);
  float* cbuf  = (float*)alloc(32 * 512 * 4);
  float* score = (float*)alloc(32 * 512 * 4);     // scores -> aw in place
  float* gates = (float*)alloc(32 * 2048 * 4);
  float* logit = (float*)alloc((size_t)32 * 32000 * 4);
  int*   maskb = (int*)alloc(32 * 4);

  float* outF   = (float*)d_out;
  float* oLps   = outF;                                    // [32,64,32000]
  float* oPred  = outF + (size_t)65536000;                 // [32,64]
  float* oState = oPred + 2048;                            // [32,64,512]
  float* oMask  = oState + (size_t)1048576;                // [32,64]

  // ---- one-time weight conversion to bf16 ----
  auto cvt = [&](const float* s, bf16* d, int n) {
    f2bf_kernel<<<(n + 255) / 256, 256, 0, stream>>>(s, d, n);
  };
  cvt(Wa,  wWa,  512 * 1024);
  cvt(Wc,  wWc,  512 * 1024);
  cvt(Wih, wWih, 2048 * 512);
  cvt(Whh, wWhh, 2048 * 512);
  cvt(Wp,  wWp,  512 * 512);
  cvt(Wh0, wWh0, 512 * 512);
  cvt(Wc0, wWc0, 512 * 512);
  cvt(Wo,  wWo,  32000 * 512);
  cvt(enc_h, ehbf, 32 * 512);
  cvt(enc_c, ecbf, 32 * 512);
  init_mask_kernel<<<1, 32, 0, stream>>>(maskb);

  // ---- h0 = enc_h @ Wh0^T (bf16 into cat1[:,512:]); c0 = enc_c @ Wc0^T ----
  gemm_bf16_32<<<8, 128, 0, stream>>>(ehbf, 512, wWh0, nullptr,
                                      nullptr, cat1 + 512, 1024, 512, 512, 0, 0);
  gemm_bf16_32<<<8, 128, 0, stream>>>(ecbf, 512, wWc0, nullptr,
                                      cbuf, nullptr, 0, 512, 512, 0, 0);

  // ---- 64 decode steps ----
  for (int t = 0; t < 64; ++t) {
    emb_kernel<<<64, 256, 0, stream>>>(embedding, trg, cat1, cat2, t);
    // scores = [emb,h] @ Wa^T + ba
    gemm_bf16_32<<<8, 128, 0, stream>>>(cat1, 1024, wWa, ba,
                                        score, nullptr, 0, 1024, 512, 0, 0);
    attn_softmax_kernel<<<32, 256, 0, stream>>>(score);
    ctx_kernel<<<64, 256, 0, stream>>>(score, enc, cat2);
    // comb = relu([emb,ctx] @ Wc^T + bc)  (bf16 only)
    gemm_bf16_32<<<8, 128, 0, stream>>>(cat2, 1024, wWc, bc,
                                        nullptr, combb, 512, 1024, 512, 1, 0);
    // gates = comb @ Wih^T + bih + h @ Whh^T + bhh
    gemm_bf16_32<<<32, 128, 0, stream>>>(combb, 512, wWih, bih,
                                         gates, nullptr, 0, 512, 2048, 0, 0);
    gemm_bf16_32<<<32, 128, 0, stream>>>(cat1 + 512, 1024, wWhh, bhh,
                                         gates, nullptr, 0, 512, 2048, 0, 1);
    // LSTM update (reads old h only via gates); writes new h_bf + states out
    lstm_kernel<<<64, 256, 0, stream>>>(gates, cbuf, cat1, oState, t);
    // pre = comb @ Wp^T + bp (bf16 only)
    gemm_bf16_32<<<8, 128, 0, stream>>>(combb, 512, wWp, bp,
                                        nullptr, prebf, 512, 512, 512, 0, 0);
    // logits = pre @ Wo^T + bo
    logits_gemm_kernel<<<500, 128, 0, stream>>>(prebf, wWo, bo, logit);
    // log_softmax, argmax, mask bookkeeping
    finalize_kernel<<<32, 256, 0, stream>>>(logit, maskb, oLps, oPred, oMask, t);
  }
}
